// GraphNetwork_59493886984402
// MI455X (gfx1250) — matrix operand order repaired and checked by reading, not compiled, not run on hardware
//
#include <hip/hip_runtime.h>

typedef __attribute__((ext_vector_type(16))) _Float16 v16h;
typedef __attribute__((ext_vector_type(8)))  _Float16 v8h;
typedef __attribute__((ext_vector_type(8)))  float    v8f;

#define WMMA_F16(a, b, c) \
  __builtin_amdgcn_wmma_f32_16x16x32_f16(false, (a), false, (b), (short)0, (c), false, false)

#if defined(__has_builtin)
#if __has_builtin(__builtin_amdgcn_sched_barrier)
#define SCHED_FENCE() __builtin_amdgcn_sched_barrier(0)
#endif
#endif
#ifndef SCHED_FENCE
#define SCHED_FENCE()
#endif

enum { MODE_PLAIN = 0, MODE_CONCAT2 = 1, MODE_GATHER2 = 2 };

// Branch-free transcendentals (v_exp_f32 + v_rcp_f32, no libm divergence).
__device__ __forceinline__ float fsigmoid(float x) {
  x = fminf(fmaxf(x, -30.0f), 30.0f);
  return __builtin_amdgcn_rcpf(1.0f + __expf(-x));
}
__device__ __forceinline__ float ftanh(float x) {
  x = fminf(fmaxf(x, -15.0f), 15.0f);
  float e = __expf(x + x);
  return (e - 1.0f) * __builtin_amdgcn_rcpf(e + 1.0f);
}

// ---------------------------------------------------------------------------
// Elementwise prep kernels
// ---------------------------------------------------------------------------
__global__ void cvt_f16_kernel(const float* __restrict__ s, _Float16* __restrict__ d, int n) {
  int i = blockIdx.x * 256 + threadIdx.x;
  if (i < n) d[i] = (_Float16)s[i];
}

__global__ void bias_sum_kernel(const float* __restrict__ a, const float* __restrict__ b,
                                float* __restrict__ o, int n) {
  int i = blockIdx.x * 256 + threadIdx.x;
  if (i < n) o[i] = a[i] + b[i];
}

// ---------------------------------------------------------------------------
// WMMA GEMM: C[M x Nout] = A[M x K](f16) @ W[Nout x K](f16)^T + bias
// Block = 128 threads (4 waves); each wave owns one 16-row M-tile and 8
// N-tiles (128 cols). grid.x = M/64, grid.y = Nout/128.
// ---------------------------------------------------------------------------
template <int MODE, bool STORE_HALF>
__global__ void gemm_wmma_kernel(const _Float16* __restrict__ A0,
                                 const _Float16* __restrict__ A1,
                                 const long long* __restrict__ edges,
                                 const _Float16* __restrict__ W,
                                 const float* __restrict__ bias,
                                 float* __restrict__ Cf,
                                 _Float16* __restrict__ Ch,
                                 int K, int Nout) {
  const int lane   = threadIdx.x & 31;
  const int wave   = threadIdx.x >> 5;
  const int half16 = lane >> 4;       // 0: lanes 0-15, 1: lanes 16-31
  const int lrow   = lane & 15;
  const int m0     = (blockIdx.x * 4 + wave) * 16;
  const int nbase  = blockIdx.y * 128;
  const int row    = m0 + lrow;       // A-row this lane loads

  long long g0 = 0, g1 = 0;
  if (MODE == MODE_GATHER2) {
    g0 = edges[2 * (long long)row];     // src
    g1 = edges[2 * (long long)row + 1]; // dst
  }

  v8f zero = {};
  v8f acc[8];
#pragma unroll
  for (int nt = 0; nt < 8; ++nt) acc[nt] = zero;

  for (int k0 = 0; k0 < K; k0 += 32) {
    // Per-lane A row base pointer at column k0 (each 32-wide K-step lies
    // entirely inside one 128-wide half for the concat/gather modes).
    const _Float16* ap;
    if (MODE == MODE_PLAIN) {
      ap = A0 + (size_t)row * K + k0;
    } else if (MODE == MODE_CONCAT2) {
      ap = (k0 < 128) ? (A0 + (size_t)row * 128 + k0)
                      : (A1 + (size_t)row * 128 + (k0 - 128));
    } else {
      ap = (k0 < 128) ? (A0 + (size_t)g0 * 128 + k0)
                      : (A0 + (size_t)g1 * 128 + (k0 - 128));
    }
    // A tile layout: lane holds M=lane%16; K = half16*16 + 0..15
    union { v16h v; v8h h[2]; } ua;
    ua.h[0] = *(const v8h*)(ap + half16 * 16);
    ua.h[1] = *(const v8h*)(ap + half16 * 16 + 8);
    v16h a = ua.v;

#pragma unroll
    for (int nt = 0; nt < 8; ++nt) {
      // B tile: b[h] = W[n][k0 + half16*16 + h], n = nbase + nt*16 + lane%16
      const _Float16* wp =
          W + (size_t)(nbase + nt * 16 + lrow) * K + k0 + half16 * 16;
      v16h b = *(const v16h*)wp;
      acc[nt] = WMMA_F16(a, b, acc[nt]);
    }
  }

  // C/D layout: c[v] -> C[m0 + v + half16*8][nbase + nt*16 + lane%16]
#pragma unroll
  for (int nt = 0; nt < 8; ++nt) {
    int ncol = nbase + nt * 16 + lrow;
    float bv = bias[ncol];
#pragma unroll
    for (int v = 0; v < 8; ++v) {
      int mr = m0 + v + half16 * 8;
      float val = acc[nt][v] + bv;
      if (STORE_HALF)
        Ch[(size_t)mr * Nout + ncol] = (_Float16)val;
      else
        Cf[(size_t)mr * Nout + ncol] = val;
    }
  }
}

// ---------------------------------------------------------------------------
// Sequential LSTM scan: one persistent workgroup, 512 threads = 16 waves.
//
// Whh rows are packed into per-wave WMMA B-tiles under a PERMUTATION chosen so
// that wave w computes exactly the 4 gate pre-activations of its own 8
// h-indices j = w*8 + 0..7:
//   permuted index p = w*32 + q,  q = nt*16 + (lane%16)
//   original row(p)  = (q>>3)*128 + w*8 + (q&7)
// => acc0[0] (q<16) holds {i|f}[j], acc1[0] (q>=16) holds {c|o}[j]; a single
// __shfl_xor(.,8) pairs them intra-wave -> no cross-wave exchange, no g LDS.
//
// h is double-buffered in LDS (read buf t&1, write buf (t+1)&1) so ONE
// barrier per step orders step-t writes against step-(t+1) reads.
// xg[t+1] is prefetched during the gate phase of step t.
// ---------------------------------------------------------------------------
__global__ void lstm_seq_kernel(const float* __restrict__ xg,   // [T, 512] (bias pre-added)
                                const float* __restrict__ Whh,  // [512, 128] f32
                                _Float16* __restrict__ nodes_h, // [T, 128] h outputs
                                int T) {
  __shared__ __align__(32) _Float16 h_buf[2][128];

  const int tid    = threadIdx.x;  // 0..511
  const int wave   = tid >> 5;     // 0..15
  const int lane   = tid & 31;
  const int half16 = lane >> 4;
  const int lrow   = lane & 15;

  // Preload permuted Whh^T tiles into registers (once).
  v16h Breg[2][4];
#pragma unroll
  for (int nt = 0; nt < 2; ++nt) {
    const int q   = nt * 16 + lrow;
    const int row = (q >> 3) * 128 + wave * 8 + (q & 7);  // gate-major permutation
    const float* wrow = Whh + (size_t)row * 128;
#pragma unroll
    for (int kk = 0; kk < 4; ++kk) {
      const float* wp = wrow + kk * 32 + half16 * 16;
      v16h b;
#pragma unroll
      for (int h = 0; h < 16; ++h) b[h] = (_Float16)wp[h];
      Breg[nt][kk] = b;
    }
  }

  if (tid < 128) {
    h_buf[0][tid] = (_Float16)0.0f;
    h_buf[1][tid] = (_Float16)0.0f;
  }

  const int j = wave * 8 + (lane & 7);  // h-index owned by lanes 0..7 of this wave
  float cst = 0.0f;                     // cell state (lanes 0..7 of each wave)
  float xvi = 0.f, xvf = 0.f, xvc = 0.f, xvo = 0.f;
  if (lane < 8) {  // software-pipelined xg, preload t=0
    xvi = xg[j];
    xvf = xg[128 + j];
    xvc = xg[256 + j];
    xvo = xg[384 + j];
  }
  __syncthreads();

  for (int t = 0; t < T; ++t) {
    const _Float16* hb = h_buf[t & 1];

    // ---- g = Whh_perm @ h : all four A chunks in one LDS clause.
    union { v16h v; v8h h[2]; } a0, a1, a2, a3;
    a0.h[0] = *(const v8h*)(hb + 0  + half16 * 16);
    a0.h[1] = *(const v8h*)(hb + 0  + half16 * 16 + 8);
    a1.h[0] = *(const v8h*)(hb + 32 + half16 * 16);
    a1.h[1] = *(const v8h*)(hb + 32 + half16 * 16 + 8);
    a2.h[0] = *(const v8h*)(hb + 64 + half16 * 16);
    a2.h[1] = *(const v8h*)(hb + 64 + half16 * 16 + 8);
    a3.h[0] = *(const v8h*)(hb + 96 + half16 * 16);
    a3.h[1] = *(const v8h*)(hb + 96 + half16 * 16 + 8);
    SCHED_FENCE();  // keep all 8 ds_loads grouped ahead of the WMMA chain

    v8f acc0 = {};
    v8f acc1 = {};
    acc0 = WMMA_F16(a0.v, Breg[0][0], acc0);
    acc1 = WMMA_F16(a0.v, Breg[1][0], acc1);
    acc0 = WMMA_F16(a1.v, Breg[0][1], acc0);
    acc1 = WMMA_F16(a1.v, Breg[1][1], acc1);
    acc0 = WMMA_F16(a2.v, Breg[0][2], acc0);
    acc1 = WMMA_F16(a2.v, Breg[1][2], acc1);
    acc0 = WMMA_F16(a3.v, Breg[0][3], acc0);
    acc1 = WMMA_F16(a3.v, Breg[1][3], acc1);

    // All D rows are identical (A rows = broadcast h), so every lane holds
    // g_perm[w*32 + (lane&15)] in acc0[0] and g_perm[w*32+16+(lane&15)] in
    // acc1[0]. Pair {i,f} and {c,o} across the xor-8 lanes.
    float g_if = acc0[0];                 // lane&7==L, lane&8==0 -> i[j]; else f[j]
    float g_co = acc1[0];                 // lane&8==0 -> c[j]; else o[j]
    float g_fi = __shfl_xor(g_if, 8);
    float g_oc = __shfl_xor(g_co, 8);

    if (lane < 8) {
      float gi = g_if + xvi;
      float gf = g_fi + xvf;
      float gc = g_co + xvc;
      float go = g_oc + xvo;

      // Prefetch xg for t+1 (clamped); hidden by gates + next matvec.
      const float* xn = xg + (size_t)((t + 1 < T) ? t + 1 : t) * 512;
      xvi = xn[j];
      xvf = xn[128 + j];
      xvc = xn[256 + j];
      xvo = xn[384 + j];

      float iv = fsigmoid(gi);
      float fv = fsigmoid(gf);
      float g2 = ftanh(gc);
      float ov = fsigmoid(go);
      cst = fv * cst + iv * g2;
      float hv = ov * ftanh(cst);
      _Float16 hh = (_Float16)hv;
      h_buf[(t + 1) & 1][j] = hh;
      nodes_h[(size_t)t * 128 + j] = hh;
    }
    __syncthreads();  // order h writes (t) against h reads (t+1)
  }
}

// ---------------------------------------------------------------------------
// Host orchestration
// ---------------------------------------------------------------------------
extern "C" void kernel_launch(void* const* d_in, const int* in_sizes, int n_in,
                              void* d_out, int out_size, void* d_ws, size_t ws_size,
                              hipStream_t stream) {
  const float*     inputs   = (const float*)d_in[0];
  const long long* edges    = (const long long*)d_in[1];  // int64 [N,2]
  const float*     in_W     = (const float*)d_in[2];
  const float*     in_b     = (const float*)d_in[3];
  const float*     out_W    = (const float*)d_in[4];
  const float*     out_b    = (const float*)d_in[5];
  const float*     edge_W   = (const float*)d_in[6];   // [2,128,256]
  const float*     edge_b   = (const float*)d_in[7];   // [2,128]
  const float*     lstm_Wih = (const float*)d_in[8];   // [2,512,256]
  const float*     lstm_Whh = (const float*)d_in[9];   // [2,512,128]
  const float*     lstm_bih = (const float*)d_in[10];  // [2,512]
  const float*     lstm_bhh = (const float*)d_in[11];  // [2,512]
  float* out = (float*)d_out;

  const int N = in_sizes[0] / 128;  // 32768

  // Workspace carve-out (256-B aligned slices).
  char*  ws  = (char*)d_ws;
  size_t off = 0;
  auto carve = [&](size_t bytes) -> void* {
    void* p = ws + off;
    off = (off + bytes + 255) & ~(size_t)255;
    return p;
  };
  _Float16* inputs_h = (_Float16*)carve((size_t)N * 128 * 2);
  _Float16* nodes_h  = (_Float16*)carve((size_t)N * 128 * 2);
  _Float16* edgeh_h  = (_Float16*)carve((size_t)N * 128 * 2);
  float*    xg       = (float*)carve((size_t)N * 512 * 4);
  _Float16* w_in     = (_Float16*)carve(128 * 128 * 2);
  _Float16* w_edge   = (_Float16*)carve(2 * 128 * 256 * 2);
  _Float16* w_ih     = (_Float16*)carve(2 * 512 * 256 * 2);
  _Float16* w_out    = (_Float16*)carve(128 * 128 * 2);
  float*    bsum     = (float*)carve(2 * 512 * 4);
  (void)ws_size; (void)n_in; (void)out_size;

  auto cvt = [&](const float* s, _Float16* d, int n) {
    cvt_f16_kernel<<<(n + 255) / 256, 256, 0, stream>>>(s, d, n);
  };
  cvt(inputs,   inputs_h, N * 128);
  cvt(in_W,     w_in,     128 * 128);
  cvt(edge_W,   w_edge,   2 * 128 * 256);
  cvt(lstm_Wih, w_ih,     2 * 512 * 256);
  cvt(out_W,    w_out,    128 * 128);
  bias_sum_kernel<<<(1024 + 255) / 256, 256, 0, stream>>>(lstm_bih, lstm_bhh, bsum, 1024);

  dim3 blk(128);
  // nodes = inputs @ in_W.T + in_b   (f16 out, gather source for edge GEMM)
  gemm_wmma_kernel<MODE_PLAIN, true><<<dim3(N / 64, 1), blk, 0, stream>>>(
      inputs_h, nullptr, nullptr, w_in, in_b, nullptr, nodes_h, 128, 128);

  for (int r = 0; r < 2; ++r) {
    // edge_h = concat(nodes[src], nodes[dst]) @ edge_W[r].T + edge_b[r]
    gemm_wmma_kernel<MODE_GATHER2, true><<<dim3(N / 64, 1), blk, 0, stream>>>(
        nodes_h, nullptr, edges, w_edge + (size_t)r * 128 * 256,
        edge_b + r * 128, nullptr, edgeh_h, 256, 128);
    // xg = concat(inputs, edge_h) @ Wih[r].T + (bih+bhh)   (f32 out)
    gemm_wmma_kernel<MODE_CONCAT2, false><<<dim3(N / 64, 4), blk, 0, stream>>>(
        inputs_h, edgeh_h, nullptr, w_ih + (size_t)r * 512 * 256,
        bsum + r * 512, xg, nullptr, 256, 512);
    // sequential scan; writes nodes_h[t] = h_t
    lstm_seq_kernel<<<1, 512, 0, stream>>>(xg, lstm_Whh + (size_t)r * 512 * 128,
                                           nodes_h, N);
  }

  // out = nodes @ out_W.T + out_b   (f32 to d_out)
  gemm_wmma_kernel<MODE_PLAIN, false><<<dim3(N / 64, 1), blk, 0, stream>>>(
      nodes_h, nullptr, nullptr, w_out, out_b, out, nullptr, 128, 128);
}
